// GSS_39848706572389
// MI455X (gfx1250) — compile-verified, matrix operand order checked by reading, not run on hardware
//
#include <hip/hip_runtime.h>
#include <hip/hip_bf16.h>

// ---------------------------------------------------------------------------
// Problem constants (match the reference)
// ---------------------------------------------------------------------------
#define T_DIM   64
#define N_DIM   100
#define NS_DIM  25
#define C_DIM   128
#define C2_DIM  256
#define ROWS_T  (N_DIM * NS_DIM)        // 2500 rows per t for vs-high
#define SLICE   (T_DIM * ROWS_T)        // 160000 elements per output slice
#define SLOPE_F 0.01f
#define BN_EPS_F 1e-5f
#define UH_TILES 7                      // ceil(100/16) row tiles for vuh

// CDNA5 async global->LDS path (guarded: falls back to plain staging)
#if defined(__has_builtin)
#if __has_builtin(__builtin_amdgcn_global_load_async_to_lds_b128) && \
    __has_builtin(__builtin_amdgcn_s_wait_asynccnt)
#define USE_ASYNC_LDS 1
#endif
#endif
#ifndef USE_ASYNC_LDS
#define USE_ASYNC_LDS 0
#endif

typedef _Float16 h16;
typedef __attribute__((ext_vector_type(16))) _Float16 v16h;
typedef __attribute__((ext_vector_type(8)))  float    v8f;
typedef int v4i_t __attribute__((ext_vector_type(4)));
typedef __attribute__((address_space(1))) v4i_t* as1_v4i;   // global int4*
typedef __attribute__((address_space(3))) v4i_t* as3_v4i;   // LDS int4*

// Exact LeakyReLU: for slope in (0,1), lrelu(x) == max(x, slope*x).
__device__ __forceinline__ float lrelu_f(float x) { return fmaxf(x, SLOPE_F * x); }

// 16-bit WMMA A/B fragment K mapping (wave32): lanes 0-15 hold K sub-bands
// {kh*8..kh*8+7, 16+kh*8..16+kh*8+7} with kh = lane>>4.
__device__ __forceinline__ int kidx_of(int khalf, int e) {
  return (e < 8) ? (khalf * 8 + e) : (16 + khalf * 8 + (e - 8));
}
// Inverse: K-in-chunk cc -> (khalf, element)
__device__ __forceinline__ void inv_kidx(int cc, int& kh, int& e) {
  if (cc < 16) { kh = cc >> 3; e = cc & 7; }
  else         { int d = cc - 16; kh = d >> 3; e = 8 + (d & 7); }
}

// ---------------------------------------------------------------------------
// Prep kernels
// ---------------------------------------------------------------------------
// Shuffle a (K x Nout) f32 weight into fragment-major f16:
//   dst[(((ct*nkk)+kk)*32 + lane)*16 + e] = W[kk*32 + kidx(lane>>4,e)][ct*16 + lane&15]
// so each lane's 16-element B fragment is one contiguous 32B block.
__global__ void k_shufW(const float* __restrict__ src, h16* __restrict__ dst,
                        int K, int Nout, int nkk) {
  int i = blockIdx.x * blockDim.x + threadIdx.x;
  if (i >= K * Nout) return;
  int e    = i & 15;
  int l    = (i >> 4) & 31;
  int frag = i >> 9;
  int kk   = frag % nkk;
  int ct   = frag / nkk;
  int k    = kk * 32 + kidx_of(l >> 4, e);
  int c    = ct * 16 + (l & 15);
  dst[i] = (h16)src[k * Nout + c];
}

__global__ void k_fold_bn(const float* __restrict__ g, const float* __restrict__ b,
                          const float* __restrict__ m, const float* __restrict__ v,
                          float* __restrict__ a, float* __restrict__ bs, int n) {
  int i = blockIdx.x * blockDim.x + threadIdx.x;
  if (i < n) {
    float av = g[i] * rsqrtf(v[i] + BN_EPS_F);
    a[i]  = av;
    bs[i] = b[i] - m[i] * av;
  }
}

// ---------------------------------------------------------------------------
// Fused VertexSimilarity (high path, C=128): d -> MLP(128->256->128->1)
// grid = (40, T), block = 128 threads (4 waves); each wave owns 16 rows.
// ---------------------------------------------------------------------------
__global__ __launch_bounds__(128) void k_vs_high(
    const float* __restrict__ nh,        // (T, N, C)
    const h16*   __restrict__ W1sw,      // shuffled (128x256) f16
    const float* __restrict__ a1, const float* __restrict__ b1,   // 256
    const h16*   __restrict__ W2sw,      // shuffled (256x128) f16
    const float* __restrict__ a2, const float* __restrict__ b2,   // 128
    const float* __restrict__ W3,        // (128)
    const float* __restrict__ b3,        // (1)
    float* __restrict__ enew,            // (T, N, NS)
    float* __restrict__ simi,            // (T, N, NS)
    float scale)
{
  __shared__ float sNodeF[NS_DIM][C_DIM];       // 12.5 KB support rows (f32)
  __shared__ h16   sDsw[4][4][32][16];          // 16 KB: d in A-frag layout / later h2
  __shared__ h16   sHsw[4][8][32][16];          // 32 KB: h1 in A-frag layout
  __shared__ float sSum[4][16];                 // per-row d sums (simi)

  const int t    = blockIdx.y;
  const int w    = threadIdx.x >> 5;
  const int lane = threadIdx.x & 31;
  const float* nhT = nh + (size_t)t * N_DIM * C_DIM;

  __builtin_prefetch(W1sw, 0, 1);
  __builtin_prefetch(W2sw, 0, 1);

  // Stage support-side node rows (rows 0..24 of this t) into LDS.
#if USE_ASYNC_LDS
  {
    char* gsrc = (char*)nhT;               // builtin takes non-const AS1 int4*
    char* ldst = (char*)&sNodeF[0][0];
    for (int i = threadIdx.x; i < (NS_DIM * C_DIM) / 4; i += 128) {
      __builtin_amdgcn_global_load_async_to_lds_b128(
          (as1_v4i)(gsrc + (size_t)i * 16),
          (as3_v4i)(ldst + (size_t)i * 16),
          0, 0);
    }
    __builtin_amdgcn_s_wait_asynccnt(0);
  }
#else
  for (int i = threadIdx.x; i < NS_DIM * C_DIM; i += 128)
    sNodeF[i / C_DIM][i % C_DIM] = nhT[i];
#endif
  if (lane < 16) sSum[w][lane] = 0.f;
  __syncthreads();

  const int rowBase = blockIdx.x * 64 + w * 16;

  // Each thread owns K-in-chunk position cc = lane; its (khalf, element) is fixed.
  int khW, eW;
  inv_kidx(lane, khW, eW);

  // Build d rows directly in A-fragment layout; accumulate row sums.
  for (int m = 0; m < 16; ++m) {
    int r  = rowBase + m;
    int rc = (r < ROWS_T) ? r : 0;
    int n  = rc / NS_DIM, s = rc - n * NS_DIM;
    float psum = 0.f;
#pragma unroll
    for (int j = 0; j < 4; ++j) {                 // j == K chunk (c = j*32 + lane)
      int c = j * 32 + lane;
      float dv = fabsf(nhT[n * C_DIM + c] - sNodeF[s][c]);
      psum += dv;
      sDsw[w][j][khW * 16 + m][eW] = (h16)dv;
    }
    atomicAdd(&sSum[w][m], psum);
  }
  __syncthreads();

  // simi_metric = -sum_c d * scale
  if (lane < 16) {
    int r = rowBase + lane;
    if (r < ROWS_T) simi[(size_t)t * ROWS_T + r] = -sSum[w][lane] * scale;
  }

  const int khalf = lane >> 4;
  const int ncol  = lane & 15;

  // --- GEMM1: (16 x 128) d @ (128 x 256) W1 -> h1 (A-frag layout in sHsw) ---
  v16h afr[4];
#pragma unroll
  for (int kk = 0; kk < 4; ++kk)
    afr[kk] = *(const v16h*)&sDsw[w][kk][lane][0];

  for (int ct = 0; ct < 16; ++ct) {
    v8f acc = {};
#pragma unroll
    for (int kk = 0; kk < 4; ++kk) {
      v16h bfr = *(const v16h*)(W1sw + (size_t)((ct * 4 + kk) * 32 + lane) * 16);
      acc = __builtin_amdgcn_wmma_f32_16x16x32_f16(false, afr[kk], false, bfr,
                                                   (short)0, acc, false, false);
    }
    int cg  = ct * 16 + ncol;
    int kk2 = cg >> 5, cc2 = cg & 31, kh2, e2;
    inv_kidx(cc2, kh2, e2);
    float sa = a1[cg], sb = b1[cg];
#pragma unroll
    for (int v = 0; v < 8; ++v) {
      int m = v + 8 * khalf;
      sHsw[w][kk2][kh2 * 16 + m][e2] = (h16)lrelu_f(acc[v] * sa + sb);
    }
  }
  __syncthreads();

  // --- GEMM2: (16 x 256) h1 @ (256 x 128) W2 -> h2 (plain rows, reuse sDsw) ---
  v16h afr2[8];
#pragma unroll
  for (int kk = 0; kk < 8; ++kk)
    afr2[kk] = *(const v16h*)&sHsw[w][kk][lane][0];

  h16* sFin = &sDsw[w][0][0][0];                 // 16 x 128 plain h2 rows
  for (int ct = 0; ct < 8; ++ct) {
    v8f acc = {};
#pragma unroll
    for (int kk = 0; kk < 8; ++kk) {
      v16h bfr = *(const v16h*)(W2sw + (size_t)((ct * 8 + kk) * 32 + lane) * 16);
      acc = __builtin_amdgcn_wmma_f32_16x16x32_f16(false, afr2[kk], false, bfr,
                                                   (short)0, acc, false, false);
    }
    int cg = ct * 16 + ncol;
    float sa = a2[cg], sb = b2[cg];
#pragma unroll
    for (int v = 0; v < 8; ++v) {
      int m = v + 8 * khalf;
      sFin[m * C_DIM + cg] = (h16)lrelu_f(acc[v] * sa + sb);
    }
  }
  __syncthreads();

  // --- Final 128->1 layer + sigmoid ---
  if (lane < 16) {
    int r = rowBase + lane;
    if (r < ROWS_T) {
      float sum = b3[0];
      for (int c = 0; c < C_DIM; ++c) sum += (float)sFin[lane * C_DIM + c] * W3[c];
      enew[(size_t)t * ROWS_T + r] = 1.f / (1.f + __expf(-sum));
    }
  }
}

// ---------------------------------------------------------------------------
// Fused VertexUpdateHigh MLP: x(256) -> 256 -> 128, WMMA.
// grid = (2, T), block = 128 (4 waves); wave w handles row tile blockIdx.x*4+w
// (clamped: tile 7 recomputes tile 6, writes are identical values).
// ---------------------------------------------------------------------------
__global__ __launch_bounds__(128) void k_vuh(
    const h16*   __restrict__ xsw,      // (T, 7, 8, 32, 16) f16 fragments
    const h16*   __restrict__ W1sw,     // shuffled (256x256) f16
    const float* __restrict__ a1, const float* __restrict__ b1,  // 256
    const h16*   __restrict__ W2sw,     // shuffled (256x128) f16
    const float* __restrict__ a2, const float* __restrict__ b2,  // 128
    float* __restrict__ nhout)          // (T, N, 128)
{
  __shared__ h16 sHsw[4][8][32][16];    // 32 KB h1 fragments

  const int t    = blockIdx.y;
  const int w    = threadIdx.x >> 5;
  const int lane = threadIdx.x & 31;
  int tile  = blockIdx.x * 4 + w;
  int tileC = (tile < UH_TILES) ? tile : (UH_TILES - 1);
  const int rowBase = tileC * 16;
  const int khalf = lane >> 4;
  const int ncol  = lane & 15;

  __builtin_prefetch(W1sw, 0, 1);

  const h16* xt = xsw + (((size_t)t * UH_TILES + tileC) * 8) * 32 * 16;

  // --- GEMM1: (16 x 256) x @ (256 x 256) W1 -> h1 ---
  v16h afr[8];
#pragma unroll
  for (int kk = 0; kk < 8; ++kk)
    afr[kk] = *(const v16h*)(xt + (size_t)(kk * 32 + lane) * 16);

  for (int ct = 0; ct < 16; ++ct) {
    v8f acc = {};
#pragma unroll
    for (int kk = 0; kk < 8; ++kk) {
      v16h bfr = *(const v16h*)(W1sw + (size_t)((ct * 8 + kk) * 32 + lane) * 16);
      acc = __builtin_amdgcn_wmma_f32_16x16x32_f16(false, afr[kk], false, bfr,
                                                   (short)0, acc, false, false);
    }
    int cg  = ct * 16 + ncol;
    int kk2 = cg >> 5, cc2 = cg & 31, kh2, e2;
    inv_kidx(cc2, kh2, e2);
    float sa = a1[cg], sb = b1[cg];
#pragma unroll
    for (int v = 0; v < 8; ++v) {
      int m = v + 8 * khalf;
      sHsw[w][kk2][kh2 * 16 + m][e2] = (h16)lrelu_f(acc[v] * sa + sb);
    }
  }
  __syncthreads();

  // --- GEMM2: (16 x 256) h1 @ (256 x 128) W2 -> nhout ---
  v16h afr2[8];
#pragma unroll
  for (int kk = 0; kk < 8; ++kk)
    afr2[kk] = *(const v16h*)&sHsw[w][kk][lane][0];

  for (int ct = 0; ct < 8; ++ct) {
    v8f acc = {};
#pragma unroll
    for (int kk = 0; kk < 8; ++kk) {
      v16h bfr = *(const v16h*)(W2sw + (size_t)((ct * 8 + kk) * 32 + lane) * 16);
      acc = __builtin_amdgcn_wmma_f32_16x16x32_f16(false, afr2[kk], false, bfr,
                                                   (short)0, acc, false, false);
    }
    int cg = ct * 16 + ncol;
    float sa = a2[cg], sb = b2[cg];
#pragma unroll
    for (int v = 0; v < 8; ++v) {
      int m  = v + 8 * khalf;
      int rr = rowBase + m;
      if (rr < N_DIM)
        nhout[((size_t)t * N_DIM + rr) * C_DIM + cg] = lrelu_f(acc[v] * sa + sb);
    }
  }
}

// ---------------------------------------------------------------------------
// Edge renormalization (shared by high/low paths). One thread per (t,n).
// ---------------------------------------------------------------------------
__global__ void k_edge(const float* __restrict__ eprev, const float* __restrict__ enw,
                       float* __restrict__ eout) {
  int i = blockIdx.x * blockDim.x + threadIdx.x;
  if (i >= T_DIM * N_DIM) return;
  int n = i % N_DIM;
  const float* ep = eprev + (size_t)i * NS_DIM;
  const float* en = enw   + (size_t)i * NS_DIM;

  float el[NS_DIM], wv[NS_DIM], ev[NS_DIM];
  float els = 0.f, wsum = 0.f, tot = 0.f;
#pragma unroll
  for (int s = 0; s < NS_DIM; ++s) {
    float msk = (n == s) ? 1.f : 0.f;
    float v = ep[s] * (1.f - msk);
    el[s] = v; els += v;
  }
#pragma unroll
  for (int s = 0; s < NS_DIM; ++s) { wv[s] = en[s] * el[s]; wsum += fabsf(wv[s]); }
  wsum = fmaxf(wsum, 1e-12f);
#pragma unroll
  for (int s = 0; s < NS_DIM; ++s) {
    float msk = (n == s) ? 1.f : 0.f;
    float v = wv[s] / wsum * els + msk + 1e-6f;
    ev[s] = v; tot += v;
  }
#pragma unroll
  for (int s = 0; s < NS_DIM; ++s) eout[(size_t)i * NS_DIM + s] = ev[s] / tot;
}

// ---------------------------------------------------------------------------
// VertexUpdateLow: lrelu(concat(eh, nl) @ W(50x25) + b). Thread per (t,n,o).
// ---------------------------------------------------------------------------
__global__ void k_vul(const float* __restrict__ eh, const float* __restrict__ nl,
                      const float* __restrict__ W, const float* __restrict__ b,
                      float* __restrict__ out) {
  int i = blockIdx.x * blockDim.x + threadIdx.x;
  if (i >= T_DIM * N_DIM * NS_DIM) return;
  int o = i % NS_DIM, tn = i / NS_DIM;
  const float* ep = eh + (size_t)tn * NS_DIM;
  const float* np = nl + (size_t)tn * NS_DIM;
  float s = b[o];
#pragma unroll
  for (int k = 0; k < NS_DIM; ++k) s += ep[k] * W[k * NS_DIM + o];
#pragma unroll
  for (int k = 0; k < NS_DIM; ++k) s += np[k] * W[(NS_DIM + k) * NS_DIM + o];
  out[i] = lrelu_f(s);
}

// ---------------------------------------------------------------------------
// VertexSimilarity low (C=25): fully scalar, one thread per (t,n,s).
// ---------------------------------------------------------------------------
__global__ void k_vs_low(const float* __restrict__ nl,
                         const float* __restrict__ W1,  // (25,50)
                         const float* __restrict__ a1, const float* __restrict__ b1,  // 50
                         const float* __restrict__ W2,  // (50,25)
                         const float* __restrict__ a2, const float* __restrict__ b2,  // 25
                         const float* __restrict__ W3,  // (25)
                         const float* __restrict__ b3,  // (1)
                         float* __restrict__ enew, float* __restrict__ simi, float scale) {
  int i = blockIdx.x * blockDim.x + threadIdx.x;
  if (i >= T_DIM * ROWS_T) return;
  int s  = i % NS_DIM;
  int tn = i / NS_DIM;
  int t  = tn / N_DIM;
  const float* av = nl + (size_t)tn * NS_DIM;
  const float* bv = nl + ((size_t)t * N_DIM + s) * NS_DIM;

  float d[NS_DIM], dsum = 0.f;
#pragma unroll
  for (int c = 0; c < NS_DIM; ++c) { d[c] = fabsf(av[c] - bv[c]); dsum += d[c]; }
  simi[i] = -dsum * scale;

  float h1[2 * NS_DIM];
#pragma unroll
  for (int o = 0; o < 2 * NS_DIM; ++o) {
    float acc = 0.f;
#pragma unroll
    for (int c = 0; c < NS_DIM; ++c) acc += d[c] * W1[c * (2 * NS_DIM) + o];
    h1[o] = lrelu_f(acc * a1[o] + b1[o]);
  }
  float lg = b3[0];
#pragma unroll
  for (int o = 0; o < NS_DIM; ++o) {
    float acc = 0.f;
#pragma unroll
    for (int c = 0; c < 2 * NS_DIM; ++c) acc += h1[c] * W2[c * NS_DIM + o];
    lg += lrelu_f(acc * a2[o] + b2[o]) * W3[o];
  }
  enew[i] = 1.f / (1.f + __expf(-lg));
}

// ---------------------------------------------------------------------------
// ef = l1-normalized off-diagonal edge_low. Thread per (t,n).
// ---------------------------------------------------------------------------
__global__ void k_ef(const float* __restrict__ el, float* __restrict__ ef) {
  int i = blockIdx.x * blockDim.x + threadIdx.x;
  if (i >= T_DIM * N_DIM) return;
  int n = i % N_DIM;
  const float* ep = el + (size_t)i * NS_DIM;
  float v[NS_DIM], sum = 0.f;
#pragma unroll
  for (int s = 0; s < NS_DIM; ++s) {
    float msk = (n == s) ? 1.f : 0.f;
    float val = ep[s] * (1.f - msk);
    v[s] = val; sum += fabsf(val);
  }
  sum = fmaxf(sum, 1e-12f);
#pragma unroll
  for (int s = 0; s < NS_DIM; ++s) ef[(size_t)i * NS_DIM + s] = v[s] / sum;
}

// ---------------------------------------------------------------------------
// aggr: rows 0..24 = ef^T @ nh (over all 100), rows 25..99 = ef @ nh[:25].
// Thread per (t,row,c).
// ---------------------------------------------------------------------------
__global__ void k_aggr(const float* __restrict__ ef, const float* __restrict__ nh,
                       float* __restrict__ aggr) {
  int i = blockIdx.x * blockDim.x + threadIdx.x;
  if (i >= T_DIM * N_DIM * C_DIM) return;
  int c   = i % C_DIM;
  int tr  = i / C_DIM;
  int row = tr % N_DIM;
  int t   = tr / N_DIM;
  const float* efT = ef + (size_t)t * N_DIM * NS_DIM;
  const float* nhT = nh + (size_t)t * N_DIM * C_DIM;
  float acc = 0.f;
  if (row < NS_DIM) {
    for (int n = 0; n < N_DIM; ++n) acc += efT[n * NS_DIM + row] * nhT[n * C_DIM + c];
  } else {
#pragma unroll
    for (int s = 0; s < NS_DIM; ++s) acc += efT[row * NS_DIM + s] * nhT[s * C_DIM + c];
  }
  aggr[i] = acc;
}

// ---------------------------------------------------------------------------
// x = concat(nh, aggr) -> f16, written directly in WMMA A-fragment layout,
// padded to 112 rows (rows >= 100 are zero). Thread per (t, rowp, cc).
// ---------------------------------------------------------------------------
__global__ void k_concat_sw(const float* __restrict__ nh, const float* __restrict__ aggr,
                            h16* __restrict__ xsw) {
  int i = blockIdx.x * blockDim.x + threadIdx.x;
  if (i >= T_DIM * (UH_TILES * 16) * C2_DIM) return;
  int cc   = i % C2_DIM;
  int tr   = i / C2_DIM;
  int rowp = tr % (UH_TILES * 16);
  int t    = tr / (UH_TILES * 16);

  float v = 0.f;
  if (rowp < N_DIM) {
    size_t tn = (size_t)t * N_DIM + rowp;
    v = (cc < C_DIM) ? nh[tn * C_DIM + cc] : aggr[tn * C_DIM + (cc - C_DIM)];
  }
  int kk = cc >> 5, c5 = cc & 31, kh, e;
  inv_kidx(c5, kh, e);
  int tile = rowp >> 4, m = rowp & 15;
  size_t idx = ((((size_t)t * UH_TILES + tile) * 8 + kk) * 32 + (kh * 16 + m)) * 16 + e;
  xsw[idx] = (h16)v;
}

// ---------------------------------------------------------------------------
// Host orchestration
// ---------------------------------------------------------------------------
extern "C" void kernel_launch(void* const* d_in, const int* in_sizes, int n_in,
                              void* d_out, int out_size, void* d_ws, size_t ws_size,
                              hipStream_t stream) {
  (void)in_sizes; (void)n_in; (void)out_size; (void)ws_size;

  const float* nh0 = (const float*)d_in[0];
  const float* nl0 = (const float*)d_in[1];
  const float* eh0 = (const float*)d_in[2];
  const float* el0 = (const float*)d_in[3];
  float* outp = (float*)d_out;

  auto P = [&](int idx) { return (const float*)d_in[idx]; };
  const int IEH = 4;
  const int GB[2] = {16, 16 + 36};

  char* wp = (char*)d_ws;
  auto alloc = [&](size_t n) -> void* {
    void* r = (void*)wp; wp += (n + 255) & ~(size_t)255; return r;
  };

  h16 *w1hHi[3], *w2hHi[3];
  float *a1Hi[3], *b1Hi[3], *a2Hi[3], *b2Hi[3];
  for (int i = 0; i < 3; ++i) {
    w1hHi[i] = (h16*)alloc(C_DIM * C2_DIM * sizeof(h16));
    w2hHi[i] = (h16*)alloc(C2_DIM * C_DIM * sizeof(h16));
    a1Hi[i] = (float*)alloc(C2_DIM * 4); b1Hi[i] = (float*)alloc(C2_DIM * 4);
    a2Hi[i] = (float*)alloc(C_DIM * 4);  b2Hi[i] = (float*)alloc(C_DIM * 4);
  }
  float *a1Lo[2], *b1Lo[2], *a2Lo[2], *b2Lo[2];
  for (int i = 0; i < 2; ++i) {
    a1Lo[i] = (float*)alloc(50 * 4); b1Lo[i] = (float*)alloc(50 * 4);
    a2Lo[i] = (float*)alloc(25 * 4); b2Lo[i] = (float*)alloc(25 * 4);
  }
  h16 *w1hUh[2], *w2hUh[2];
  float *a1Uh[2], *b1Uh[2], *a2Uh[2], *b2Uh[2];
  for (int i = 0; i < 2; ++i) {
    w1hUh[i] = (h16*)alloc(C2_DIM * C2_DIM * sizeof(h16));
    w2hUh[i] = (h16*)alloc(C2_DIM * C_DIM * sizeof(h16));
    a1Uh[i] = (float*)alloc(C2_DIM * 4); b1Uh[i] = (float*)alloc(C2_DIM * 4);
    a2Uh[i] = (float*)alloc(C_DIM * 4);  b2Uh[i] = (float*)alloc(C_DIM * 4);
  }
  float* enewBuf  = (float*)alloc((size_t)SLICE * 4);
  float* simiDmy  = (float*)alloc((size_t)SLICE * 4);
  float* ehPP[2]  = {(float*)alloc((size_t)SLICE * 4), (float*)alloc((size_t)SLICE * 4)};
  float* elPP[2]  = {(float*)alloc((size_t)SLICE * 4), (float*)alloc((size_t)SLICE * 4)};
  float* nlPP[2]  = {(float*)alloc((size_t)SLICE * 4), (float*)alloc((size_t)SLICE * 4)};
  float* efBuf    = (float*)alloc((size_t)SLICE * 4);
  float* nhPP[2]  = {(float*)alloc((size_t)T_DIM * N_DIM * C_DIM * 4),
                     (float*)alloc((size_t)T_DIM * N_DIM * C_DIM * 4)};
  float* aggrBuf  = (float*)alloc((size_t)T_DIM * N_DIM * C_DIM * 4);
  h16*   xswBuf   = (h16*)alloc((size_t)T_DIM * UH_TILES * 16 * C2_DIM * sizeof(h16));

  // ---- prep: shuffle weights to fragment layout, fold BN ----
  auto prep_vs_high = [&](int base, int i) {
    k_shufW<<<(C_DIM * C2_DIM + 255) / 256, 256, 0, stream>>>(P(base + 0), w1hHi[i], C_DIM, C2_DIM, 4);
    k_shufW<<<(C2_DIM * C_DIM + 255) / 256, 256, 0, stream>>>(P(base + 5), w2hHi[i], C2_DIM, C_DIM, 8);
    k_fold_bn<<<1, 256, 0, stream>>>(P(base + 1), P(base + 2), P(base + 3), P(base + 4), a1Hi[i], b1Hi[i], C2_DIM);
    k_fold_bn<<<1, 128, 0, stream>>>(P(base + 6), P(base + 7), P(base + 8), P(base + 9), a2Hi[i], b2Hi[i], C_DIM);
  };
  prep_vs_high(IEH, 0);
  prep_vs_high(GB[0], 1);
  prep_vs_high(GB[1], 2);
  for (int g = 0; g < 2; ++g) {
    int bl = GB[g] + 14;  // vsl
    k_fold_bn<<<1, 64, 0, stream>>>(P(bl + 1), P(bl + 2), P(bl + 3), P(bl + 4), a1Lo[g], b1Lo[g], 50);
    k_fold_bn<<<1, 32, 0, stream>>>(P(bl + 6), P(bl + 7), P(bl + 8), P(bl + 9), a2Lo[g], b2Lo[g], 25);
    int bu = GB[g] + 26;  // vuh
    k_shufW<<<(C2_DIM * C2_DIM + 255) / 256, 256, 0, stream>>>(P(bu + 0), w1hUh[g], C2_DIM, C2_DIM, 8);
    k_shufW<<<(C2_DIM * C_DIM + 255) / 256, 256, 0, stream>>>(P(bu + 5), w2hUh[g], C2_DIM, C_DIM, 8);
    k_fold_bn<<<1, 256, 0, stream>>>(P(bu + 1), P(bu + 2), P(bu + 3), P(bu + 4), a1Uh[g], b1Uh[g], C2_DIM);
    k_fold_bn<<<1, 128, 0, stream>>>(P(bu + 6), P(bu + 7), P(bu + 8), P(bu + 9), a2Uh[g], b2Uh[g], C_DIM);
  }

  const dim3 vsGrid(40, T_DIM), vsBlk(128);
  const dim3 uhGrid(2, T_DIM);
  const int tnB = (T_DIM * N_DIM + 255) / 256;
  const int slB = (SLICE + 255) / 256;

  // ---- ieh: initial high-edge pass ----
  k_vs_high<<<vsGrid, vsBlk, 0, stream>>>(nh0, w1hHi[0], a1Hi[0], b1Hi[0],
                                          w2hHi[0], a2Hi[0], b2Hi[0],
                                          P(IEH + 10), P(IEH + 11),
                                          enewBuf, simiDmy, 1.0f);
  k_edge<<<tnB, 256, 0, stream>>>(eh0, enewBuf, ehPP[0]);

  const float* curNH = nh0;
  const float* curNL = nl0;
  const float* curEH = ehPP[0];
  const float* curEL = el0;

  for (int g = 0; g < 2; ++g) {
    const int b  = GB[g];
    const int bl = b + 14, bu = b + 26;
    const int vi = 1 + g;

    // vsh -> eh, simi_high
    k_vs_high<<<vsGrid, vsBlk, 0, stream>>>(curNH, w1hHi[vi], a1Hi[vi], b1Hi[vi],
                                            w2hHi[vi], a2Hi[vi], b2Hi[vi],
                                            P(b + 10), P(b + 11),
                                            enewBuf, outp + (size_t)(1 * 2 + g) * SLICE, 1.0f);
    float* ehNext = (g == 0) ? ehPP[1] : ehPP[0];
    k_edge<<<tnB, 256, 0, stream>>>(curEH, enewBuf, ehNext);
    curEH = ehNext;
    (void)hipMemcpyAsync(outp + (size_t)(0 * 2 + g) * SLICE, curEH,
                         (size_t)SLICE * 4, hipMemcpyDeviceToDevice, stream);

    // vul -> nl
    k_vul<<<slB, 256, 0, stream>>>(curEH, curNL, P(b + 12), P(b + 13), nlPP[g]);
    curNL = nlPP[g];

    // vsl -> el, simi_low
    k_vs_low<<<slB, 256, 0, stream>>>(curNL, P(bl + 0), a1Lo[g], b1Lo[g],
                                      P(bl + 5), a2Lo[g], b2Lo[g],
                                      P(bl + 10), P(bl + 11),
                                      enewBuf, outp + (size_t)(3 * 2 + g) * SLICE, 1.0f);
    float* elNext = elPP[g];
    k_edge<<<tnB, 256, 0, stream>>>(curEL, enewBuf, elNext);
    curEL = elNext;
    (void)hipMemcpyAsync(outp + (size_t)(2 * 2 + g) * SLICE, curEL,
                         (size_t)SLICE * 4, hipMemcpyDeviceToDevice, stream);

    // vuh -> nh
    k_ef<<<tnB, 256, 0, stream>>>(curEL, efBuf);
    k_aggr<<<(T_DIM * N_DIM * C_DIM + 255) / 256, 256, 0, stream>>>(efBuf, curNH, aggrBuf);
    k_concat_sw<<<(T_DIM * UH_TILES * 16 * C2_DIM + 255) / 256, 256, 0, stream>>>(curNH, aggrBuf, xswBuf);
    k_vuh<<<uhGrid, vsBlk, 0, stream>>>(xswBuf, w1hUh[g], a1Uh[g], b1Uh[g],
                                        w2hUh[g], a2Uh[g], b2Uh[g], nhPP[g]);
    curNH = nhPP[g];
  }
}
